// LoRALinear_22359599743386
// MI455X (gfx1250) — matrix operand order repaired and checked by reading, not compiled, not run on hardware
//
#include <hip/hip_runtime.h>

typedef __bf16 bf16;
typedef __attribute__((ext_vector_type(16))) __bf16 v16bf;
typedef __attribute__((ext_vector_type(8)))  __bf16 v8bf;
typedef __attribute__((ext_vector_type(8)))  float  v8f;
typedef __attribute__((ext_vector_type(4)))  float  v4f;
typedef __attribute__((ext_vector_type(4)))  unsigned u32x4;
typedef __attribute__((ext_vector_type(8)))  int      i32x8;
typedef __attribute__((ext_vector_type(4)))  int      i32x4;

#define B_     8
#define N_     7057
#define IN_    1024
#define OUT_   1024
#define RANK_  64
#define P_     84
#define M_TOT  (B_ * N_)          // 56456
#define YMAX   (336 * 336)        // worst-case upsampled pixels (rate 4)
#define YBATCH ((long)YMAX * RANK_)

__device__ __forceinline__ float cubic_w(float t) {
    const float a = -0.75f;
    float at = fabsf(t);
    float w1 = ((a + 2.0f) * at - (a + 3.0f)) * at * at + 1.0f;
    float w2 = (((at - 5.0f) * at + 8.0f) * at - 4.0f) * a;
    return at <= 1.0f ? w1 : (at < 2.0f ? w2 : 0.0f);
}

__device__ __forceinline__ int clampi(int v, int lo, int hi) {
    return v < lo ? lo : (v > hi ? hi : v);
}

__device__ __forceinline__ int rate_of(int e) { return e == 0 ? 2 : (e == 1 ? 1 : 4); }

// ---------------------------------------------------------------------------
// TDM: DMA one 2-D bf16 tile (tile_d1 rows x tile_d0 elems) from a row-major
// matrix (row pitch = stride0 elems) into LDS at lds_byte_addr.
// LDS padding: +16B after every 64B stored -> row pitch 80B (= 40 bf16).
// Rows beyond tensor_d1 are zero-filled (OOB semantics).
// D# packing per CDNA5 ISA.
// 6-arg builtin form (this toolchain): (g0, g1, g2, g3, extra, cpol);
// groups 2/3 + extra are unused for <=2-D tiles -> zero-filled.
// ---------------------------------------------------------------------------
__device__ __forceinline__ void tdm_load_tile(
    unsigned lds_byte_addr, const void* gptr,
    unsigned tensor_d0, unsigned tensor_d1,
    unsigned tile_d0, unsigned tile_d1, unsigned stride0)
{
    unsigned long ga = (unsigned long)gptr;
    u32x4 g0;
    g0[0] = 1u;                                   // count=1 (valid), user mode
    g0[1] = lds_byte_addr;                        // lds_addr [63:32]
    g0[2] = (unsigned)ga;                         // global_addr [95:64]
    g0[3] = (unsigned)((ga >> 32) & 0x01FFFFFFu)  // global_addr [120:96]
          | 0x80000000u;                          // type=2 ("image") [127:126]
    i32x8 g1;
    g1[0] = (1 << 16)      // data_size = 2 bytes
          | (1 << 20)      // pad_enable
          | (3 << 22)      // pad_interval: pad after 64 bytes
          | (3 << 25);     // pad_amount: 4 dwords (16 bytes)
    g1[1] = (int)((tensor_d0 & 0xFFFFu) << 16);                       // td0[15:0]
    g1[2] = (int)((tensor_d0 >> 16) | ((tensor_d1 & 0xFFFFu) << 16)); // td0[31:16]|td1[15:0]
    g1[3] = (int)((tensor_d1 >> 16) | (tile_d0 << 16));               // td1[31:16]|tile0
    g1[4] = (int)tile_d1;                                             // tile1 | tile2=0
    g1[5] = (int)stride0;                                             // stride0[31:0]
    g1[6] = 0;                                                        // stride0 hi, stride1 lo
    g1[7] = 0;
    i32x4 gz4 = {0, 0, 0, 0};
    i32x8 gz8 = {0, 0, 0, 0, 0, 0, 0, 0};
    __builtin_amdgcn_tensor_load_to_lds(g0, g1, gz4, gz4, gz8, 0);
}

// ---------------------------------------------------------------------------
// fp32 -> bf16 bulk convert (n multiple of 8)
// ---------------------------------------------------------------------------
__global__ __launch_bounds__(256) void cvt_f32_bf16(
    const float* __restrict__ src, bf16* __restrict__ dst, long n)
{
    long i = ((long)blockIdx.x * 256 + threadIdx.x) * 8;
    if (i >= n) return;
    v4f a = *(const v4f*)(src + i);
    v4f b = *(const v4f*)(src + i + 4);
    bf16 o[8];
#pragma unroll
    for (int j = 0; j < 4; ++j) { o[j] = (bf16)a[j]; o[4 + j] = (bf16)b[j]; }
    *(v8bf*)(dst + i) = *(v8bf*)o;
}

// ---------------------------------------------------------------------------
// bf16 WMMA GEMM with TDM double-buffered tile pipeline:
//   C[M,N] = A[M,K] * B[N,K]^T  (+bias) (+=C)   A,B bf16; C fp32 or bf16
// BM=128, BK=32.  8 waves: 4 in M (32 rows each), 2 in N (BN/2 cols each).
// ---------------------------------------------------------------------------
template <int BN, bool ADD_BIAS, bool ACCUM, bool OUT_BF16>
__global__ __launch_bounds__(256) void gemm_tdm_wmma(
    const bf16* __restrict__ A, const bf16* __restrict__ Bm,
    const float* __restrict__ bias, void* __restrict__ Cp,
    int M, int N, int K)
{
    constexpr int BM  = 128;
    constexpr int BK  = 32;
    constexpr int LDT = 40;             // TDM-padded row pitch (bf16 units)
    constexpr int FN  = BN / 32;
    constexpr int ASZ = BM * LDT;       // bf16 elems per A buffer
    constexpr int BSZ = BN * LDT;

    __shared__ bf16 sm[2 * ASZ + 2 * BSZ];

    const int tid  = threadIdx.x;
    const int lane = tid & 31;
    const int wid  = tid >> 5;
    const int m0   = blockIdx.y * BM;
    const int n0   = blockIdx.x * BN;

    v8f acc[2][FN];
#pragma unroll
    for (int i = 0; i < 2; ++i)
#pragma unroll
        for (int j = 0; j < FN; ++j)
            acc[i][j] = v8f{};

    const int KT = K / BK;
    const unsigned smbase = (unsigned)(unsigned long)(uintptr_t)sm;   // LDS byte offset

    auto issue = [&](int kt) {
        const int buf = kt & 1;
        tdm_load_tile(smbase + (unsigned)(buf * ASZ) * 2u,
                      A + (long)m0 * K + (long)kt * BK,
                      (unsigned)K, (unsigned)(M - m0), BK, BM, (unsigned)K);
        tdm_load_tile(smbase + (unsigned)(2 * ASZ + buf * BSZ) * 2u,
                      Bm + (long)n0 * K + (long)kt * BK,
                      (unsigned)K, (unsigned)(N - n0), BK, BN, (unsigned)K);
    };

    if (wid == 0) {
        issue(0);
        if (KT > 1) issue(1);
    }

    const int half = (lane < 16) ? 0 : 1;
    const int lr   = lane & 15;
    const int wm   = (wid & 3) * 32;            // wave M offset
    const int wn   = (wid >> 2) * (BN / 2);     // wave N offset

    for (int kt = 0; kt < KT; ++kt) {
        if (wid == 0) {   // current tile's 2 DMAs done; next tile's may be in flight
            if (kt + 1 < KT) __builtin_amdgcn_s_wait_tensorcnt(2);
            else             __builtin_amdgcn_s_wait_tensorcnt(0);
        }
        __syncthreads();

        const bf16* Ab = sm + (kt & 1) * ASZ;
        const bf16* Bb = sm + 2 * ASZ + (kt & 1) * BSZ;

        // A fragments: lane<16 holds K 0..7 & 16..23; lane>=16 holds K 8..15 & 24..31
        v16bf afr[2];
#pragma unroll
        for (int fm = 0; fm < 2; ++fm) {
            const bf16* pa = &Ab[(wm + fm * 16 + lr) * LDT + half * 8];
            union { v16bf v; v8bf h[2]; } u;
            u.h[0] = *(const v8bf*)(pa);
            u.h[1] = *(const v8bf*)(pa + 16);
            afr[fm] = u.v;
        }
        // B fragments: lane<16 holds K 0..7 & 16..23; lane>=16 holds K 8..15 & 24..31 (N = lane%16)
        v16bf bfr[FN];
#pragma unroll
        for (int fn = 0; fn < FN; ++fn) {
            const bf16* pb = &Bb[(wn + fn * 16 + lr) * LDT + half * 8];
            union { v16bf v; v8bf h[2]; } u;
            u.h[0] = *(const v8bf*)(pb);
            u.h[1] = *(const v8bf*)(pb + 16);
            bfr[fn] = u.v;
        }
#pragma unroll
        for (int fm = 0; fm < 2; ++fm)
#pragma unroll
            for (int fn = 0; fn < FN; ++fn)
                acc[fm][fn] = __builtin_amdgcn_wmma_f32_16x16x32_bf16(
                    false, afr[fm], false, bfr[fn], (short)0, acc[fm][fn],
                    false, false);

        __syncthreads();
        if (wid == 0 && kt + 2 < KT) issue(kt + 2);
    }

    // C layout per ISA: VGPR r -> M = r (lanes 0-15) / 8+r (lanes 16-31); N = lane%16
    float* Cf = (float*)Cp;
    bf16*  Cb = (bf16*)Cp;
#pragma unroll
    for (int fm = 0; fm < 2; ++fm)
#pragma unroll
        for (int fn = 0; fn < FN; ++fn)
#pragma unroll
            for (int r = 0; r < 8; ++r) {
                int gm = m0 + wm + fm * 16 + r + (lane < 16 ? 0 : 8);
                int gn = n0 + wn + fn * 16 + lr;
                if (gm < M) {
                    float v = acc[fm][fn][r];
                    if (ADD_BIAS) v += bias[gn];
                    long idx = (long)gm * N + gn;
                    if (OUT_BF16) {
                        Cb[idx] = (bf16)v;
                    } else {
                        if (ACCUM) v += Cf[idx];
                        Cf[idx] = v;
                    }
                }
            }
}

// ---------------------------------------------------------------------------
// Per-batch mean pool over the 7056 spatial tokens + gate argmax
// ---------------------------------------------------------------------------
__global__ __launch_bounds__(64) void pool_gate(
    const bf16* __restrict__ z, const float* __restrict__ W_gate,
    const float* __restrict__ b_gate, int* __restrict__ gidx)
{
    const int b = blockIdx.x;
    const int r = threadIdx.x;   // 0..63 channel
    const bf16* zb = z + ((long)b * N_ + 1) * RANK_;
    float s = 0.0f;
    for (int n = 0; n < N_ - 1; ++n) s += (float)zb[(long)n * RANK_ + r];
    __shared__ float pool[RANK_];
    __shared__ float sc[3];
    pool[r] = s * (1.0f / (float)(N_ - 1));
    __syncthreads();
    if (r < 3) {
        float a = b_gate[r];
        for (int k = 0; k < RANK_; ++k) a += pool[k] * W_gate[r * RANK_ + k];
        sc[r] = a;
    }
    __syncthreads();
    if (r == 0) {
        int   bi = 0;
        float bv = sc[0];
        if (sc[1] > bv) { bv = sc[1]; bi = 1; }
        if (sc[2] > bv) { bv = sc[2]; bi = 2; }
        gidx[b] = bi;
    }
}

// ---------------------------------------------------------------------------
// conv_w (3,64,64,3,3) -> wT[e][i][tap][o]  for coalesced channel dots
// ---------------------------------------------------------------------------
__global__ __launch_bounds__(256) void transpose_w(
    const float* __restrict__ cw, float* __restrict__ wT)
{
    int idx = blockIdx.x * 256 + threadIdx.x;      // 3*64*64*9 = 110592 exact
    int tap = idx % 9;
    int t   = idx / 9;
    int i   = t % 64; t /= 64;
    int o   = t % 64;
    int e   = t / 64;
    wT[(((long)e * 64 + i) * 9 + tap) * 64 + o] = cw[idx];
}

// ---------------------------------------------------------------------------
// Bicubic upsample of the selected expert's input: feat(64,84,84) -> y bf16
// channel-last layout y[b][pix][i] for coalesced access
// ---------------------------------------------------------------------------
__global__ __launch_bounds__(256) void upsample_bicubic(
    const bf16* __restrict__ z, const int* __restrict__ gidx,
    bf16* __restrict__ y)
{
    const int b = blockIdx.y;
    const int r = rate_of(gidx[b]);
    const int H2 = P_ * r;
    long idx = (long)blockIdx.x * 256 + threadIdx.x;
    long total = (long)H2 * H2 * RANK_;
    if (idx >= total) return;

    int  i   = (int)(idx & 63);
    long pix = idx >> 6;
    int  ow  = (int)(pix % H2);
    int  oh  = (int)(pix / H2);

    float inv = 1.0f / (float)r;
    float sy = (oh + 0.5f) * inv - 0.5f;
    float sx = (ow + 0.5f) * inv - 0.5f;
    int iy0 = (int)floorf(sy);
    int ix0 = (int)floorf(sx);

    float wy[4], wx[4];
    int   py[4], px[4];
#pragma unroll
    for (int t = 0; t < 4; ++t) {
        int pyy = iy0 - 1 + t;
        int pxx = ix0 - 1 + t;
        wy[t] = cubic_w(sy - (float)pyy);
        wx[t] = cubic_w(sx - (float)pxx);
        py[t] = clampi(pyy, 0, P_ - 1);
        px[t] = clampi(pxx, 0, P_ - 1);
    }

    const bf16* zb = z + ((long)b * N_ + 1) * RANK_;
    float acc = 0.0f;
#pragma unroll
    for (int ty = 0; ty < 4; ++ty)
#pragma unroll
        for (int tx = 0; tx < 4; ++tx)
            acc += wy[ty] * wx[tx] * (float)zb[((long)py[ty] * P_ + px[tx]) * RANK_ + i];

    y[(long)b * YBATCH + pix * RANK_ + i] = (bf16)acc;
}

// ---------------------------------------------------------------------------
// Fused 3x3 conv (zero pad) + bias + bicubic downsample, factored:
//   S[tap][i] = sum_{4x4 down taps} dw * y[i, ...]     (shared across o)
//   out[o]    = conv_b[o] + sum_{i,tap} wT[e][i][tap][o] * S[tap][i]
// writes directly into lora buffer (== z buffer, bf16), scaled by 1/RANK
// ---------------------------------------------------------------------------
__global__ __launch_bounds__(256) void conv_down(
    const bf16* __restrict__ y, const float* __restrict__ wT,
    const float* __restrict__ conv_b, const int* __restrict__ gidx,
    bf16* __restrict__ lora)
{
    __shared__ float S[9 * RANK_];
    __shared__ float part[4 * RANK_];

    const int b = blockIdx.y;
    const int p = blockIdx.x;              // 0..7055
    const int h = p / P_, w = p % P_;
    const int e = gidx[b];
    const int r = rate_of(e);
    const int H2 = P_ * r;

    float sy = (h + 0.5f) * (float)r - 0.5f;
    float sx = (w + 0.5f) * (float)r - 0.5f;
    int iy0 = (int)floorf(sy);
    int ix0 = (int)floorf(sx);

    float dwy[4], dwx[4];
    int   py[4], px[4];
#pragma unroll
    for (int t = 0; t < 4; ++t) {
        int pyy = iy0 - 1 + t;
        int pxx = ix0 - 1 + t;
        dwy[t] = cubic_w(sy - (float)pyy);
        dwx[t] = cubic_w(sx - (float)pxx);
        py[t] = clampi(pyy, 0, H2 - 1);
        px[t] = clampi(pxx, 0, H2 - 1);
    }

    const bf16* yb = y + (long)b * YBATCH;
    const int tid = threadIdx.x;

    // phase 1: build S (576 entries)
    for (int s = tid; s < 9 * RANK_; s += 256) {
        int tap = s >> 6;          // 0..8
        int i   = s & 63;
        int dy  = tap / 3 - 1;
        int dx  = tap % 3 - 1;
        float acc = 0.0f;
#pragma unroll
        for (int ty = 0; ty < 4; ++ty) {
            int ry = py[ty] + dy;
            if ((unsigned)ry >= (unsigned)H2) continue;   // conv zero-pad
#pragma unroll
            for (int tx = 0; tx < 4; ++tx) {
                int rx = px[tx] + dx;
                if ((unsigned)rx >= (unsigned)H2) continue;
                acc += dwy[ty] * dwx[tx] *
                       (float)yb[((long)ry * H2 + rx) * RANK_ + i];
            }
        }
        S[s] = acc;
    }
    __syncthreads();

    // phase 2: channel dot products, split 4 ways over i
    {
        int o  = tid & 63;
        int pt = tid >> 6;
        float acc = 0.0f;
        for (int ii = pt * 16; ii < pt * 16 + 16; ++ii)
#pragma unroll
            for (int tap = 0; tap < 9; ++tap)
                acc += wT[(((long)e * 64 + ii) * 9 + tap) * 64 + o] * S[tap * 64 + ii];
        part[pt * 64 + o] = acc;
    }
    __syncthreads();

    if (tid < 64) {
        float v = conv_b[e * 64 + tid] +
                  part[tid] + part[64 + tid] + part[128 + tid] + part[192 + tid];
        lora[((long)b * N_ + 1 + p) * RANK_ + tid] = (bf16)(v * (1.0f / (float)RANK_));
    }
}

// ---------------------------------------------------------------------------
extern "C" void kernel_launch(void* const* d_in, const int* in_sizes, int n_in,
                              void* d_out, int out_size, void* d_ws, size_t ws_size,
                              hipStream_t stream)
{
    const float* x      = (const float*)d_in[0];
    const float* W0     = (const float*)d_in[1];
    const float* b0     = (const float*)d_in[2];
    const float* W_down = (const float*)d_in[3];
    const float* W_gate = (const float*)d_in[4];
    const float* b_gate = (const float*)d_in[5];
    const float* conv_w = (const float*)d_in[6];
    const float* conv_b = (const float*)d_in[7];
    const float* W_up   = (const float*)d_in[8];
    float* out = (float*)d_out;

    // workspace layout (y aliases xb: xb dead after the two x-GEMMs)
    char* ws = (char*)d_ws;
    size_t off = 0;
    auto take = [&](size_t bytes) {
        char* p = ws + off;
        off = (off + bytes + 255) & ~(size_t)255;
        return p;
    };
    bf16* xb  = (bf16*)take((size_t)M_TOT * IN_ * 2);        // 115.6 MB (also y)
    bf16* y   = xb;
    bf16* W0b = (bf16*)take((size_t)OUT_ * IN_ * 2);         // 2 MB
    bf16* Wdb = (bf16*)take((size_t)RANK_ * IN_ * 2);        // 128 KB
    bf16* Wub = (bf16*)take((size_t)OUT_ * RANK_ * 2);       // 128 KB
    bf16* zb  = (bf16*)take((size_t)M_TOT * RANK_ * 2);      // 7.2 MB (also lora)
    int*  gidx = (int*)take(256);
    float* wT  = (float*)take((size_t)3 * 64 * 64 * 9 * 4);  // 442 KB

    // 0) bulk fp32 -> bf16 conversion (enables TDM + halves GEMM traffic)
    {
        long n = (long)M_TOT * IN_;
        cvt_f32_bf16<<<(unsigned)(n / 2048), 256, 0, stream>>>(x, xb, n);
        cvt_f32_bf16<<<(OUT_ * IN_) / 2048, 256, 0, stream>>>(W0, W0b, OUT_ * IN_);
        cvt_f32_bf16<<<(RANK_ * IN_) / 2048, 256, 0, stream>>>(W_down, Wdb, RANK_ * IN_);
        cvt_f32_bf16<<<(OUT_ * RANK_) / 2048, 256, 0, stream>>>(W_up, Wub, OUT_ * RANK_);
    }
    // 1) out = x @ W0^T + b0      (TDM + bf16 WMMA, M=56456, N=1024, K=1024)
    {
        dim3 grid(OUT_ / 128, (M_TOT + 127) / 128);
        gemm_tdm_wmma<128, true, false, false><<<grid, 256, 0, stream>>>(
            xb, W0b, b0, out, M_TOT, OUT_, IN_);
    }
    // 2) z = x @ W_down^T          (M=56456, N=64, K=1024) -> bf16
    {
        dim3 grid(1, (M_TOT + 127) / 128);
        gemm_tdm_wmma<64, false, false, true><<<grid, 256, 0, stream>>>(
            xb, Wdb, nullptr, zb, M_TOT, RANK_, IN_);
    }
    // 3) gate: pooled mean + argmax
    pool_gate<<<B_, 64, 0, stream>>>(zb, W_gate, b_gate, gidx);
    // 4) weight transpose for coalesced conv dots
    transpose_w<<<(3 * 64 * 64 * 9) / 256, 256, 0, stream>>>(conv_w, wT);
    // 5) bicubic upsample of selected expert input (worst-case grid, early exit)
    {
        dim3 grid((YMAX * RANK_) / 256, B_);
        upsample_bicubic<<<grid, 256, 0, stream>>>(zb, gidx, y);
    }
    // 6) fused conv3x3 + bias + bicubic downsample -> lora (in zb buffer), scaled
    {
        dim3 grid(P_ * P_, B_);
        conv_down<<<grid, 256, 0, stream>>>(y, wT, conv_b, gidx, zb);
    }
    // 7) out += lora @ W_up^T      (M=56456, N=1024, K=64)
    {
        dim3 grid(OUT_ / 128, (M_TOT + 127) / 128);
        gemm_tdm_wmma<128, false, true, false><<<grid, 256, 0, stream>>>(
            zb, Wub, nullptr, out, M_TOT, OUT_, RANK_);
    }
}
